// LayerNormTreeGRUCell_61366492725517
// MI455X (gfx1250) — matrix operand order repaired and checked by reading, not compiled, not run on hardware
//
#include <hip/hip_runtime.h>
#include <hip/hip_bf16.h>

// ---------------------------------------------------------------------------
// LayerNorm Tree-GRU cell, fused for MI455X (gfx1250, wave32, WMMA).
//   I = H = 512, N = 65536 children.
// Single pass over the 128 MB `hiddens` tensor. bf16 WMMA GEMM with f32
// accumulation; slab converted to bf16 once at staging (no VALU in the WMMA
// loop); B fragments double-buffered in registers to hide L2 latency.
// 512-thread blocks (16 waves x 32 h-cols) keep VGPR use < 256 so no
// s_set_vgpr_msb traffic and 4 waves/SIMD occupancy.
// ---------------------------------------------------------------------------

typedef __attribute__((ext_vector_type(16))) __bf16          v16bf;
typedef __attribute__((ext_vector_type(8)))  float           v8f;
typedef __attribute__((ext_vector_type(16))) unsigned short  v16us;

#define HDIM   512
#define IDIM   512
#define NCH    65536
#define K1_WGS 512     // workgroups for main kernel
#define SLABS  4       // slabs of MROWS rows per workgroup
#define MROWS  32      // rows per slab
#define LDSTRB 520     // bf16 slab row stride (shorts): conflict-free A reads

__device__ __forceinline__ unsigned short f32_bf16(float f) {
  unsigned int u = __float_as_uint(f);
  unsigned int r = u + 0x7FFFu + ((u >> 16) & 1u);   // round-to-nearest-even
  return (unsigned short)(r >> 16);
}

__device__ __forceinline__ float bf16_f32(unsigned short s) {
  return __uint_as_float(((unsigned int)s) << 16);
}

// --------------------------- prep kernels ---------------------------------

__global__ void k_prep_wrh(const float* __restrict__ wrh,
                           unsigned short* __restrict__ wrh_bf) {
  int idx = (blockIdx.x * blockDim.x + threadIdx.x) * 4;  // 256*256*4 = 262144
  float4 v = *reinterpret_cast<const float4*>(wrh + idx);
  wrh_bf[idx + 0] = f32_bf16(v.x);
  wrh_bf[idx + 1] = f32_bf16(v.y);
  wrh_bf[idx + 2] = f32_bf16(v.z);
  wrh_bf[idx + 3] = f32_bf16(v.w);
}

__global__ void k_prep_xri(const float* __restrict__ x,
                           const float* __restrict__ Wri,
                           float* __restrict__ xri) {
  __shared__ float xs[IDIM];
  int tid = threadIdx.x;                      // 256 threads
  xs[tid]       = x[tid];
  xs[tid + 256] = x[tid + 256];
  __syncthreads();
  for (int j = tid; j < HDIM; j += 256) {
    const float* wr = Wri + (size_t)j * IDIM;
    float s = 0.f;
    for (int i = 0; i < IDIM; ++i) s += xs[i] * wr[i];
    xri[j] = s;
  }
}

// ------------------------------ main kernel -------------------------------
// 16 waves/WG. Wave w owns h-columns [w*32, w*32+32) (2 N-tiles of 16).
// Slab = 32 rows (2 M-tiles), stored bf16 in LDS. acc[mt][t] = WMMA f32 accs.

__global__ void __launch_bounds__(512) k_main(
    const float* __restrict__ hiddens,
    const unsigned short* __restrict__ wrh_bf,
    const float* __restrict__ xri,
    const float* __restrict__ gr, const float* __restrict__ br,
    float* __restrict__ hsum_part, float* __restrict__ rhsum_part)
{
  __shared__ unsigned short slab_bf[MROWS * LDSTRB];  // 33,280 B
  __shared__ float psum[16][32];
  __shared__ float psq [16][32];
  __shared__ float muS  [32];
  __shared__ float rstdS[32];
  __shared__ float rhacc[HDIM];
  __shared__ float hsbuf[4 * HDIM];

  const int tid   = threadIdx.x;
  const int lane  = tid & 31;
  const int wave  = tid >> 5;
  const int hi    = lane >> 4;      // 0/1 half of wave
  const int ln    = lane & 15;
  const int wbase = wave * 32;

  // loop-invariant per-lane gate params at this lane's 2 h positions
  float gr_t[2], br_t[2], xri_t[2];
#pragma unroll
  for (int t = 0; t < 2; ++t) {
    int h = wbase + t * 16 + ln;
    gr_t[t] = gr[h]; br_t[t] = br[h]; xri_t[t] = xri[h];
  }

  // loop-invariant B base pointers (h row of Wrh, k-chunk picked by lane half)
  const unsigned short* bptr[2];
#pragma unroll
  for (int t = 0; t < 2; ++t)
    bptr[t] = wrh_bf + (size_t)(wbase + t * 16 + ln) * HDIM + hi * 8;
  const int aoff = 8 * hi;  // A layout: lanes<16 -> K 0..7/16..23, else +8

  rhacc[tid] = 0.f;
  float hs[4] = {0.f, 0.f, 0.f, 0.f};   // hidden_sum partials (f32, staged)
  __syncthreads();

#define LOADB(dst, KS)                                                        \
  do {                                                                        \
    _Pragma("unroll")                                                         \
    for (int t = 0; t < 2; ++t) {                                             \
      const uint4* p = reinterpret_cast<const uint4*>(bptr[t] + (KS) * 32);   \
      dst[t][0] = p[0]; dst[t][1] = p[2];                                     \
    }                                                                         \
  } while (0)

#define LOADA(af, KS)                                                         \
  do {                                                                        \
    _Pragma("unroll")                                                         \
    for (int mt = 0; mt < 2; ++mt) {                                          \
      const uint4* p = reinterpret_cast<const uint4*>(                        \
          &slab_bf[(mt * 16 + ln) * LDSTRB + (KS) * 32 + aoff]);              \
      union { v16us v; uint4 q[2]; } ua;                                      \
      ua.q[0] = p[0]; ua.q[1] = p[2]; /* shorts 0..7 and 16..23 */            \
      af[mt] = __builtin_bit_cast(v16bf, ua.v);                               \
    }                                                                         \
  } while (0)

#define DOWMMA(af, bb)                                                        \
  do {                                                                        \
    _Pragma("unroll")                                                         \
    for (int t = 0; t < 2; ++t) {                                             \
      union { v16us v; uint4 q[2]; } ub;                                      \
      ub.q[0] = bb[t][0]; ub.q[1] = bb[t][1];                                 \
      v16bf bf = __builtin_bit_cast(v16bf, ub.v);                             \
      _Pragma("unroll")                                                       \
      for (int mt = 0; mt < 2; ++mt)                                          \
        acc[mt][t] = __builtin_amdgcn_wmma_f32_16x16x32_bf16(                 \
            false, af[mt], false, bf, (short)0, acc[mt][t], false, false);    \
    }                                                                         \
  } while (0)

  for (int it = 0; it < SLABS; ++it) {
    const long long row0 = (long long)(blockIdx.x + it * K1_WGS) * MROWS;

    // ---- stage slab: f32 -> bf16 once; fold hidden_sum from exact f32 ----
#pragma unroll
    for (int i = 0; i < 8; ++i) {
      int idx = (tid + i * 512) * 4;
      int rr = idx >> 9, cc = idx & 511;
      float4 v = *reinterpret_cast<const float4*>(hiddens + row0 * HDIM + idx);
      hs[0] += v.x; hs[1] += v.y; hs[2] += v.z; hs[3] += v.w;
      uint2 pv;
      pv.x = f32_bf16(v.x) | ((unsigned)f32_bf16(v.y) << 16);
      pv.y = f32_bf16(v.z) | ((unsigned)f32_bf16(v.w) << 16);
      *reinterpret_cast<uint2*>(&slab_bf[rr * LDSTRB + cc]) = pv;
    }
    __syncthreads();

    // warm caches for the next slab while this one computes
    if (it + 1 < SLABS) {
      const float* nxt = hiddens + (row0 + (long long)K1_WGS * MROWS) * HDIM;
#pragma unroll
      for (int i = 0; i < 8; ++i)
        __builtin_prefetch(nxt + (tid + i * 512) * 4, 0, 3);
    }

    // ---- GEMM: C[m,h] = sum_k hiddens[m,k] * Wrh[h,k] via bf16 WMMA ----
    v8f acc[2][2];
#pragma unroll
    for (int mt = 0; mt < 2; ++mt)
#pragma unroll
      for (int t = 0; t < 2; ++t)
#pragma unroll
        for (int e = 0; e < 8; ++e) acc[mt][t][e] = 0.f;

    uint4 bA[2][2], bB[2][2];
    v16bf af[2];
    LOADB(bA, 0);
#pragma unroll 1
    for (int ks = 0; ks < 16; ks += 2) {
      LOADB(bB, ks + 1);            // prefetch next B while WMMAs run
      LOADA(af, ks);
      DOWMMA(af, bA);
      LOADB(bA, (ks + 2) & 15);     // wraps harmlessly on last iteration
      LOADA(af, ks + 1);
      DOWMMA(af, bB);
    }

    // ---- row statistics for LayerNorm (include x@Wri.T bias) ----
#pragma unroll
    for (int mt = 0; mt < 2; ++mt) {
#pragma unroll
      for (int r = 0; r < 8; ++r) {
        float s1 = 0.f, s2 = 0.f;
#pragma unroll
        for (int t = 0; t < 2; ++t) {
          float v = acc[mt][t][r] + xri_t[t];
          s1 += v; s2 += v * v;
        }
#pragma unroll
        for (int m = 1; m < 16; m <<= 1) {
          s1 += __shfl_xor(s1, m, 32);
          s2 += __shfl_xor(s2, m, 32);
        }
        int mrow = mt * 16 + r + (hi << 3);
        if (ln == 0) { psum[wave][mrow] = s1; psq[wave][mrow] = s2; }
      }
    }
    __syncthreads();

    if (tid < 32) {
      float s1 = 0.f, s2 = 0.f;
#pragma unroll
      for (int w = 0; w < 16; ++w) { s1 += psum[w][tid]; s2 += psq[w][tid]; }
      float mu = s1 * (1.f / HDIM);
      float var = s2 * (1.f / HDIM) - mu * mu;
      muS[tid] = mu;
      rstdS[tid] = rsqrtf(var + 1e-5f);
    }
    __syncthreads();

    // ---- r = sigmoid(LN(...)); accumulate (r * hiddens) over rows ----
#pragma unroll
    for (int t = 0; t < 2; ++t) {
      float rhp = 0.f;
      int hcol = wbase + t * 16 + ln;
#pragma unroll
      for (int mt = 0; mt < 2; ++mt)
#pragma unroll
        for (int r = 0; r < 8; ++r) {
          int mrow = mt * 16 + r + (hi << 3);
          float v  = acc[mt][t][r] + xri_t[t];
          float rn = (v - muS[mrow]) * rstdS[mrow] * gr_t[t] + br_t[t];
          float rg = 1.f / (1.f + expf(-rn));
          rhp += rg * bf16_f32(slab_bf[mrow * LDSTRB + hcol]);
        }
      rhp += __shfl_xor(rhp, 16, 32);            // fold the two row halves
      if (lane < 16) rhacc[hcol] += rhp;         // exclusive h-owner: no race
    }
    __syncthreads();                              // slab reuse next iteration
  }

  // ---- combine the four per-column hidden_sum owners deterministically ----
  {
    int o = tid >> 7;                 // column-owner group id (0..3)
    int c = (tid * 4) & 511;          // this thread's fixed 4-column base
    hsbuf[o * HDIM + c + 0] = hs[0];
    hsbuf[o * HDIM + c + 1] = hs[1];
    hsbuf[o * HDIM + c + 2] = hs[2];
    hsbuf[o * HDIM + c + 3] = hs[3];
  }
  __syncthreads();

  {
    int h = tid;                      // 512 threads: one h each
    float s = hsbuf[h] + hsbuf[HDIM + h] + hsbuf[2 * HDIM + h] +
              hsbuf[3 * HDIM + h];
    hsum_part [(size_t)blockIdx.x * HDIM + h] = s;
    rhsum_part[(size_t)blockIdx.x * HDIM + h] = rhacc[h];
  }
}

// --------------------- deterministic partial reduction --------------------

__global__ void k_reduce(const float* __restrict__ hsum_part,
                         const float* __restrict__ rhsum_part,
                         float* __restrict__ hidden_sum,
                         float* __restrict__ rh_sum) {
  int h = blockIdx.x * blockDim.x + threadIdx.x;   // 2 x 256 = 512
  float a = 0.f, b = 0.f;
  for (int w = 0; w < K1_WGS; ++w) {
    a += hsum_part [(size_t)w * HDIM + h];
    b += rhsum_part[(size_t)w * HDIM + h];
  }
  hidden_sum[h] = a;
  rh_sum[h]     = b;
}

// ----------------------- final gates (z, u) + output ----------------------

__global__ void __launch_bounds__(512) k_final(
    const float* __restrict__ x,
    const float* __restrict__ Wz, const float* __restrict__ Wu,
    const float* __restrict__ gz, const float* __restrict__ bz,
    const float* __restrict__ gu, const float* __restrict__ bu,
    const float* __restrict__ hidden_sum, const float* __restrict__ rh_sum,
    float* __restrict__ out)
{
  __shared__ float cz[IDIM + HDIM];
  __shared__ float cu[IDIM + HDIM];
  __shared__ float zarr[HDIM];
  __shared__ float uarr[HDIM];
  int tid = threadIdx.x;
  cz[tid] = x[tid];              cu[tid] = x[tid];
  cz[IDIM + tid] = hidden_sum[tid];
  cu[IDIM + tid] = rh_sum[tid];
  __syncthreads();

  const float* wzr = Wz + (size_t)tid * (IDIM + HDIM);
  const float* wur = Wu + (size_t)tid * (IDIM + HDIM);
  float zp = 0.f, up = 0.f;
  for (int j = 0; j < IDIM + HDIM; ++j) { zp += cz[j] * wzr[j]; up += cu[j] * wur[j]; }
  zarr[tid] = zp; uarr[tid] = up;
  __syncthreads();

  float s1z = 0.f, s2z = 0.f, s1u = 0.f, s2u = 0.f;
  for (int j = 0; j < HDIM; ++j) {
    float a = zarr[j]; s1z += a; s2z += a * a;
    float c = uarr[j]; s1u += c; s2u += c * c;
  }
  float muz = s1z * (1.f / HDIM), varz = s2z * (1.f / HDIM) - muz * muz;
  float muu = s1u * (1.f / HDIM), varu = s2u * (1.f / HDIM) - muu * muu;
  float zn = (zp - muz) * rsqrtf(varz + 1e-5f) * gz[tid] + bz[tid];
  float un = (up - muu) * rsqrtf(varu + 1e-5f) * gu[tid] + bu[tid];
  float z = 1.f / (1.f + expf(-zn));
  float u = tanhf(un);
  out[tid] = z * hidden_sum[tid] + (1.f - z) * u;
}

// ------------------------------- launcher ---------------------------------

extern "C" void kernel_launch(void* const* d_in, const int* in_sizes, int n_in,
                              void* d_out, int out_size, void* d_ws, size_t ws_size,
                              hipStream_t stream) {
  const float* x       = (const float*)d_in[0];
  const float* hiddens = (const float*)d_in[1];
  const float* Wz      = (const float*)d_in[2];
  const float* Wri     = (const float*)d_in[3];
  const float* Wrh     = (const float*)d_in[4];
  const float* Wu      = (const float*)d_in[5];
  const float* gz      = (const float*)d_in[6];
  const float* bz      = (const float*)d_in[7];
  const float* gr      = (const float*)d_in[8];
  const float* br      = (const float*)d_in[9];
  const float* gu      = (const float*)d_in[10];
  const float* bu      = (const float*)d_in[11];
  float* out = (float*)d_out;

  char* ws = (char*)d_ws;
  unsigned short* wrh_bf = (unsigned short*)ws;                       // 512 KB
  float* xri        = (float*)(ws + 524288);                          // 2 KB
  float* hsum_part  = (float*)(ws + 526336);                          // 1 MB
  float* rhsum_part = (float*)(ws + 526336 + 1048576);                // 1 MB
  float* hidden_sum = (float*)(ws + 526336 + 2097152);                // 2 KB
  float* rh_sum     = (float*)(ws + 526336 + 2097152 + 2048);         // 2 KB

  k_prep_wrh<<<256, 256, 0, stream>>>(Wrh, wrh_bf);
  k_prep_xri<<<1, 256, 0, stream>>>(x, Wri, xri);
  k_main<<<K1_WGS, 512, 0, stream>>>(hiddens, wrh_bf, xri, gr, br,
                                     hsum_part, rhsum_part);
  k_reduce<<<2, 256, 0, stream>>>(hsum_part, rhsum_part, hidden_sum, rh_sum);
  k_final<<<1, 512, 0, stream>>>(x, Wz, Wu, gz, bz, gu, bu,
                                 hidden_sum, rh_sum, out);
}
